// GCN_GAT_53498112639138
// MI455X (gfx1250) — compile-verified, matrix operand order checked by reading, not run on hardware
//
#include <hip/hip_runtime.h>
#include <hip/hip_bf16.h>

// ---------------------------------------------------------------------------
// Types for CDNA5 WMMA (wave32, 16x16x32 bf16 -> f32 accumulate)
// ---------------------------------------------------------------------------
typedef __attribute__((ext_vector_type(16))) __bf16 v16bf;
typedef __attribute__((ext_vector_type(8)))  __bf16 v8bf;
typedef __attribute__((ext_vector_type(8)))  float  v8f;

union AFrag { v16bf v; struct { v8bf lo, hi; } s; };

#if defined(__HIP_DEVICE_COMPILE__) && __has_builtin(__builtin_amdgcn_global_load_async_to_lds_b128)
#define HAVE_ASYNC_LDS 1
#else
#define HAVE_ASYNC_LDS 0
#endif

#if HAVE_ASYNC_LDS
// Builtin signature (from clang diagnostic): param0 is
// 'int __attribute__((vector_size(16))) __device__ *'  i.e. v4i addrspace(1)*,
// param1 the LDS-side v4i addrspace(3)*.
typedef int v4i_t __attribute__((vector_size(16)));
typedef __attribute__((address_space(1))) v4i_t g_v4i;
typedef __attribute__((address_space(3))) v4i_t l_v4i;
#endif

#define TPB 256
#define NEG_SLOPE 0.2f
#define BN_EPS 1e-5f

static __device__ __forceinline__ float lrelu(float v) {
    return v > 0.0f ? v : NEG_SLOPE * v;
}

static __device__ __forceinline__ void atomicMaxF(float* addr, float val) {
    if (val >= 0.0f) atomicMax((int*)addr, __float_as_int(val));
    else             atomicMin((unsigned int*)addr, __float_as_uint(val));
}

// ---------------------------------------------------------------------------
// Elementwise helpers
// ---------------------------------------------------------------------------
__global__ void k_fill(float* p, float v, long long n) {
    long long i = (long long)blockIdx.x * blockDim.x + threadIdx.x;
    if (i < n) p[i] = v;
}

__global__ void k_deg(const int* __restrict__ dst, float* deg, int E) {
    int e = blockIdx.x * blockDim.x + threadIdx.x;
    if (e < E) atomicAdd(&deg[dst[e]], 1.0f);
}

__global__ void k_rsqrt_inplace(float* p, int n) {
    int i = blockIdx.x * blockDim.x + threadIdx.x;
    if (i < n) p[i] = rsqrtf(p[i]);
}

// ---------------------------------------------------------------------------
// A staging: A[M,K] f32 -> Ab[M,KP] bf16, KP = 32-aligned K, zero padded.
// ---------------------------------------------------------------------------
__global__ void k_prepA(const float* __restrict__ A, __bf16* __restrict__ Ab,
                        int M, int K, int KP) {
    long long i = (long long)blockIdx.x * blockDim.x + threadIdx.x;
    if (i >= (long long)M * KP) return;
    int m = (int)(i / KP), k = (int)(i % KP);
    Ab[i] = (__bf16)((k < K) ? A[(long long)m * K + k] : 0.0f);
}

// ---------------------------------------------------------------------------
// Weight staging: W[K,Nout] f32 (row-major) -> Wb in B-fragment layout:
//   Wb[ct][kc][lane][16] bf16, zero padded -> one unconditional 32B load.
// ---------------------------------------------------------------------------
__global__ void k_prepB(const float* __restrict__ W, __bf16* __restrict__ Wb,
                        int K, int Nout, int KC, int CT) {
    int idx = blockIdx.x * blockDim.x + threadIdx.x;
    int tot = CT * KC * 512;
    if (idx >= tot) return;
    int j    = idx & 15;
    int lane = (idx >> 4) & 31;
    int t    = idx >> 9;
    int kc   = t % KC;
    int ct   = t / KC;
    int col  = ct * 16 + (lane & 15);
    int koff = (lane >> 4) * 8;
    int k    = (kc << 5) + (j < 8 ? (koff + j) : (16 + koff + (j - 8)));
    float v  = (k < K && col < Nout) ? W[(long long)k * Nout + col] : 0.0f;
    Wb[idx]  = (__bf16)v;
}

// ---------------------------------------------------------------------------
// WMMA GEMM: C[M,Nout] = Ab[M,KP](bf16) @ Wb (+bias, +relu)
// 8 waves -> 64x128 block tile; each wave: 2x2 WMMA tiles (4 v_wmma/K-step).
// A tile double-buffered in LDS (one barrier per K-step); staging is a pure
// 16B global->LDS copy (async-to-LDS when the toolchain exposes it).
// ---------------------------------------------------------------------------
__global__ void k_gemm_wmma(const __bf16* __restrict__ Ab,
                            const __bf16* __restrict__ Wb,
                            float* __restrict__ C,
                            int M, int KP, int Nout, int KC, int CT,
                            const float* __restrict__ bias, int relu) {
    __shared__ __bf16 As[2][64][40];   // stride 80B: every v8bf slot 16B aligned

    const int tid  = threadIdx.x;
    const int lane = tid & 31;
    const int wave = tid >> 5;
    const int waveM = wave >> 2;          // 0..1
    const int waveN = wave & 3;           // 0..3
    const int blockM = blockIdx.x * 64;
    const int ct0 = blockIdx.y * 8 + waveN * 2;
    const int row16 = lane & 15;
    const int koff  = (lane >> 4) * 8;    // 0 or 8

    const int ldRow  = tid >> 2;          // 0..63
    const int ldKseg = (tid & 3) * 8;     // 0,8,16,24
    long long arow = blockM + ldRow;
    if (arow >= M) arow = M - 1;          // clamp: padded rows never stored
    const __bf16* asrc0 = Ab + arow * KP + ldKseg;

    v8f acc00 = {}, acc01 = {}, acc10 = {}, acc11 = {};

    for (int kc = 0; kc < KC; ++kc) {
        const int p = kc & 1;
        const __bf16* asrc = asrc0 + (kc << 5);

        // ---- stage 16B of A into LDS (pure copy; no cvt, no guards) ----
#if HAVE_ASYNC_LDS
        __builtin_amdgcn_global_load_async_to_lds_b128(
            (g_v4i*)(unsigned long long)asrc,
            (l_v4i*)(unsigned)(unsigned long long)&As[p][ldRow][ldKseg],
            0, 0);
#if __has_builtin(__builtin_amdgcn_s_wait_asynccnt)
        __builtin_amdgcn_s_wait_asynccnt(0);
#else
        asm volatile("s_wait_asynccnt 0" ::: "memory");
#endif
#else
        *(v8bf*)&As[p][ldRow][ldKseg] = *(const v8bf*)asrc;
#endif
        __syncthreads();

        // ---- B fragments: pre-staged, one 32B vector load per tile ----
        v16bf b0 = {}, b1 = {};
        if (ct0 < CT)
            b0 = *(const v16bf*)(Wb + (((long long)ct0 * KC + kc) * 32 + lane) * 16);
        if (ct0 + 1 < CT)
            b1 = *(const v16bf*)(Wb + (((long long)(ct0 + 1) * KC + kc) * 32 + lane) * 16);

        // ---- A fragments from LDS (two aligned 16B loads each) ----
        const int rl0 = waveM * 32 + row16;
        AFrag a0, a1;
        a0.s.lo = *(const v8bf*)&As[p][rl0][koff];
        a0.s.hi = *(const v8bf*)&As[p][rl0][16 + koff];
        a1.s.lo = *(const v8bf*)&As[p][rl0 + 16][koff];
        a1.s.hi = *(const v8bf*)&As[p][rl0 + 16][16 + koff];

        acc00 = __builtin_amdgcn_wmma_f32_16x16x32_bf16(false, a0.v, false, b0, (short)0, acc00, false, false);
        acc01 = __builtin_amdgcn_wmma_f32_16x16x32_bf16(false, a0.v, false, b1, (short)0, acc01, false, false);
        acc10 = __builtin_amdgcn_wmma_f32_16x16x32_bf16(false, a1.v, false, b0, (short)0, acc10, false, false);
        acc11 = __builtin_amdgcn_wmma_f32_16x16x32_bf16(false, a1.v, false, b1, (short)0, acc11, false, false);
        // Double-buffered LDS: the single barrier above also orders next
        // iteration's overwrite of buffer p^1 against the previous iteration's reads.
    }

    // ---- store: VGPR r -> row r / r+8 per half-wave, N = lane&15 ----
    v8f accs[2][2] = { { acc00, acc01 }, { acc10, acc11 } };
    #pragma unroll
    for (int mi = 0; mi < 2; ++mi) {
        int rbase = blockM + waveM * 32 + mi * 16 + ((lane >> 4) << 3);
        #pragma unroll
        for (int ni = 0; ni < 2; ++ni) {
            int col = (ct0 + ni) * 16 + row16;
            if (col < Nout) {
                float badd = bias ? bias[col] : 0.0f;
                #pragma unroll
                for (int r = 0; r < 8; ++r) {
                    int rr = rbase + r;
                    if (rr < M) {
                        float v = accs[mi][ni][r] + badd;
                        if (relu) v = fmaxf(v, 0.0f);
                        C[(long long)rr * Nout + col] = v;
                    }
                }
            }
        }
    }
}

// ---------------------------------------------------------------------------
// GCN: edge scatter  agg[dst] += dinv[src]*dinv[dst]*h[src]   (wave per edge)
// ---------------------------------------------------------------------------
__global__ void k_gcn_edge(const int* __restrict__ src, const int* __restrict__ dst,
                           const float* __restrict__ dinv,
                           const float* __restrict__ h, float* __restrict__ agg,
                           int E, int F) {
    int wid  = (blockIdx.x * blockDim.x + threadIdx.x) >> 5;
    int lane = threadIdx.x & 31;
    if (wid >= E) return;
    int s = src[wid], d = dst[wid];
    float c = dinv[s] * dinv[d];
    const float* hs = h + (long long)s * F;
    float* ad = agg + (long long)d * F;
    __builtin_prefetch(hs, 0, 0);
    for (int f = lane; f < F; f += 32) atomicAdd(&ad[f], c * hs[f]);
}

// out = BN(ReLU(agg + dinv^2*h + b))
__global__ void k_gcn_final(const float* __restrict__ agg, const float* __restrict__ h,
                            const float* __restrict__ dinv, const float* __restrict__ gb,
                            const float* __restrict__ bg, const float* __restrict__ bb,
                            const float* __restrict__ bm, const float* __restrict__ bv,
                            float* __restrict__ out, int N, int F) {
    long long i = (long long)blockIdx.x * blockDim.x + threadIdx.x;
    if (i >= (long long)N * F) return;
    int n = (int)(i / F), f = (int)(i % F);
    float di = dinv[n];
    float v = agg[i] + di * di * h[i] + gb[f];
    v = fmaxf(v, 0.0f);
    v = (v - bm[f]) * rsqrtf(bv[f] + BN_EPS) * bg[f] + bb[f];
    out[i] = v;
}

// ---------------------------------------------------------------------------
// GAT kernels (H = 3 heads, C = per-head width, power of two -> shiftC)
// ---------------------------------------------------------------------------
__global__ void k_gat_logits(const float* __restrict__ h,
                             const float* __restrict__ a_s, const float* __restrict__ a_d,
                             float* __restrict__ als, float* __restrict__ ald,
                             int N, int C) {
    int idx = blockIdx.x * blockDim.x + threadIdx.x;   // (node, head)
    if (idx >= N * 3) return;
    int n = idx / 3, hh = idx % 3;
    const float* hr  = h + (long long)n * 3 * C + hh * C;
    const float* asr = a_s + hh * C;
    const float* adr = a_d + hh * C;
    float ss = 0.0f, sd = 0.0f;
    for (int c = 0; c < C; ++c) { float v = hr[c]; ss += v * asr[c]; sd += v * adr[c]; }
    als[idx] = ss; ald[idx] = sd;
}

__global__ void k_gat_selfmax(const float* __restrict__ als, const float* __restrict__ ald,
                              float* __restrict__ mmax, int N3) {
    int i = blockIdx.x * blockDim.x + threadIdx.x;
    if (i < N3) mmax[i] = lrelu(als[i] + ald[i]);
}

__global__ void k_gat_edgemax(const int* __restrict__ src, const int* __restrict__ dst,
                              const float* __restrict__ als, const float* __restrict__ ald,
                              float* __restrict__ mmax, int E) {
    int idx = blockIdx.x * blockDim.x + threadIdx.x;   // (edge, head)
    if (idx >= E * 3) return;
    int e = idx / 3, hh = idx % 3;
    int s = src[e], d = dst[e];
    atomicMaxF(&mmax[d * 3 + hh], lrelu(als[s * 3 + hh] + ald[d * 3 + hh]));
}

__global__ void k_gat_wsinit(const float* __restrict__ als, const float* __restrict__ ald,
                             const float* __restrict__ mmax, float* __restrict__ denom,
                             int N3) {
    int i = blockIdx.x * blockDim.x + threadIdx.x;
    if (i < N3) denom[i] = __expf(lrelu(als[i] + ald[i]) - mmax[i]);
}

__global__ void k_gat_outinit(const float* __restrict__ h, const float* __restrict__ denom,
                              float* __restrict__ out, int N, int F, int shiftC) {
    long long i = (long long)blockIdx.x * blockDim.x + threadIdx.x;
    if (i >= (long long)N * F) return;
    int n = (int)(i / F), f = (int)(i % F);
    out[i] = denom[n * 3 + (f >> shiftC)] * h[i];
}

__global__ void k_gat_edge(const int* __restrict__ src, const int* __restrict__ dst,
                           const float* __restrict__ als, const float* __restrict__ ald,
                           const float* __restrict__ mmax, float* __restrict__ denom,
                           const float* __restrict__ h, float* __restrict__ out,
                           int E, int F, int shiftC) {
    int wid  = (blockIdx.x * blockDim.x + threadIdx.x) >> 5;
    int lane = threadIdx.x & 31;
    if (wid >= E) return;
    int s = src[wid], d = dst[wid];
    float we[3];
    #pragma unroll
    for (int hh = 0; hh < 3; ++hh)
        we[hh] = __expf(lrelu(als[s * 3 + hh] + ald[d * 3 + hh]) - mmax[d * 3 + hh]);
    if (lane < 3) atomicAdd(&denom[d * 3 + lane], we[lane]);
    const float* hs = h + (long long)s * F;
    float* od = out + (long long)d * F;
    __builtin_prefetch(hs, 0, 0);
    for (int f = lane; f < F; f += 32)
        atomicAdd(&od[f], we[f >> shiftC] * hs[f]);
}

__global__ void k_gat_final(float* __restrict__ out, const float* __restrict__ denom,
                            const float* __restrict__ ab,
                            const float* __restrict__ bg, const float* __restrict__ bb,
                            const float* __restrict__ bm, const float* __restrict__ bv,
                            int N, int F, int shiftC) {
    long long i = (long long)blockIdx.x * blockDim.x + threadIdx.x;
    if (i >= (long long)N * F) return;
    int n = (int)(i / F), f = (int)(i % F);
    float v = out[i] / denom[n * 3 + (f >> shiftC)] + ab[f];
    v = fmaxf(v, 0.0f);
    v = (v - bm[f]) * rsqrtf(bv[f] + BN_EPS) * bg[f] + bb[f];
    out[i] = v;
}

__global__ void k_pool(const float* __restrict__ xin, const int* __restrict__ batch,
                       float* __restrict__ cat, int N, int F, int stride, int colOff) {
    long long i = (long long)blockIdx.x * blockDim.x + threadIdx.x;
    if (i >= (long long)N * F) return;
    int n = (int)(i / F), f = (int)(i % F);
    atomicAdd(&cat[(long long)batch[n] * stride + colOff + f], xin[i]);
}

// ---------------------------------------------------------------------------
// Launch
// ---------------------------------------------------------------------------
static inline int cdiv(long long a, long long b) { return (int)((a + b - 1) / b); }

static void gemm(const float* A, const float* W, float* C,
                 int M, int K, int Nout, const float* bias, int relu,
                 __bf16* wsW, __bf16* wsA, hipStream_t stream) {
    int KC = cdiv(K, 32), CT = cdiv(Nout, 16), KP = KC * 32;
    k_prepA<<<cdiv((long long)M * KP, TPB), TPB, 0, stream>>>(A, wsA, M, K, KP);
    k_prepB<<<cdiv((long long)CT * KC * 512, TPB), TPB, 0, stream>>>(W, wsW, K, Nout, KC, CT);
    dim3 g(cdiv(M, 64), cdiv(Nout, 128));
    k_gemm_wmma<<<g, TPB, 0, stream>>>(wsA, wsW, C, M, KP, Nout, KC, CT, bias, relu);
}

extern "C" void kernel_launch(void* const* d_in, const int* in_sizes, int n_in,
                              void* d_out, int out_size, void* d_ws, size_t ws_size,
                              hipStream_t stream) {
    const int N = in_sizes[2];
    const int E = in_sizes[1] / 2;
    const int G = out_size;

    const float* x     = (const float*)d_in[0];
    const int*   ei    = (const int*)d_in[1];
    const int*   src   = ei;
    const int*   dst   = ei + E;
    const int*   batch = (const int*)d_in[2];

    const float* gw[3] = { (const float*)d_in[3], (const float*)d_in[5], (const float*)d_in[7] };
    const float* gb[3] = { (const float*)d_in[4], (const float*)d_in[6], (const float*)d_in[8] };

    const float* bn[6][4];
    for (int i = 0; i < 6; ++i)
        for (int j = 0; j < 4; ++j)
            bn[i][j] = (const float*)d_in[9 + i * 4 + j];

    const float* aw[3]  = { (const float*)d_in[33], (const float*)d_in[37], (const float*)d_in[41] };
    const float* as_[3] = { (const float*)d_in[34], (const float*)d_in[38], (const float*)d_in[42] };
    const float* ad_[3] = { (const float*)d_in[35], (const float*)d_in[39], (const float*)d_in[43] };
    const float* ab_[3] = { (const float*)d_in[36], (const float*)d_in[40], (const float*)d_in[44] };

    const float* f1w = (const float*)d_in[45]; const float* f1b = (const float*)d_in[46];
    const float* f2w = (const float*)d_in[47]; const float* f2b = (const float*)d_in[48];
    const float* f3w = (const float*)d_in[49]; const float* f3b = (const float*)d_in[50];

    // Workspace carve-up
    float* ws    = (float*)d_ws;
    float* buf1  = ws;                              // N*768
    float* buf2  = buf1 + (long long)N * 768;       // N*768
    float* dinv  = buf2 + (long long)N * 768;       // N
    float* als   = dinv + N;                        // N*3
    float* ald   = als + (long long)N * 3;          // N*3
    float* mmax  = ald + (long long)N * 3;          // N*3
    float* denom = mmax + (long long)N * 3;         // N*3
    float* cat   = denom + (long long)N * 3;        // G*512
    float* fc1   = cat + (long long)G * 512;        // G*256
    float* fc2   = fc1 + (long long)G * 256;        // G*64
    // bf16 staging areas, 32B aligned
    unsigned long long pbase = (unsigned long long)(fc2 + (long long)G * 64);
    pbase = (pbase + 31ull) & ~31ull;
    __bf16* wsW = (__bf16*)pbase;                   // <=0.6M bf16 (weights)
    pbase += (1ull << 21);                          // 2 MB
    pbase = (pbase + 31ull) & ~31ull;
    __bf16* wsA = (__bf16*)pbase;                   // N*768 bf16 (activations)
    (void)ws_size; (void)n_in;

    const long long N3 = (long long)N * 3;

    // --- degrees: deg = 1 + count(dst); dinv = rsqrt(deg)
    k_fill<<<cdiv(N, TPB), TPB, 0, stream>>>(dinv, 1.0f, N);
    k_deg<<<cdiv(E, TPB), TPB, 0, stream>>>(dst, dinv, E);
    k_rsqrt_inplace<<<cdiv(N, TPB), TPB, 0, stream>>>(dinv, N);

    // =================== GCN tower ===================
    const int gcnF[3] = { 256, 128, 128 };
    const int gcnK[3] = { 373, 256, 128 };
    const float* gcnIn[3];
    float* gcnH[3];  float* gcnAgg[3];
    gcnIn[0] = x;    gcnH[0] = buf1; gcnAgg[0] = buf2;
    gcnIn[1] = buf1; gcnH[1] = buf2; gcnAgg[1] = buf1;
    gcnIn[2] = buf2; gcnH[2] = buf1; gcnAgg[2] = buf2;

    for (int L = 0; L < 3; ++L) {
        int F = gcnF[L], K = gcnK[L];
        gemm(gcnIn[L], gw[L], gcnH[L], N, K, F, nullptr, 0, wsW, wsA, stream);
        k_fill<<<cdiv((long long)N * F, TPB), TPB, 0, stream>>>(gcnAgg[L], 0.0f, (long long)N * F);
        k_gcn_edge<<<cdiv((long long)E * 32, TPB), TPB, 0, stream>>>(src, dst, dinv, gcnH[L], gcnAgg[L], E, F);
        k_gcn_final<<<cdiv((long long)N * F, TPB), TPB, 0, stream>>>(
            gcnAgg[L], gcnH[L], dinv, gb[L],
            bn[L][0], bn[L][1], bn[L][2], bn[L][3],
            (float*)gcnH[L], N, F);
    }
    k_fill<<<cdiv((long long)G * 512, TPB), TPB, 0, stream>>>(cat, 0.0f, (long long)G * 512);
    k_pool<<<cdiv((long long)N * 128, TPB), TPB, 0, stream>>>(buf1, batch, cat, N, 128, 512, 0);

    // =================== GAT tower ===================
    const int gatF[3]  = { 768, 384, 384 };
    const int gatK[3]  = { 373, 768, 384 };
    const int gatC[3]  = { 256, 128, 128 };
    const int gatSh[3] = { 8, 7, 7 };
    const float* gatIn = x;

    for (int L = 0; L < 3; ++L) {
        int F = gatF[L], K = gatK[L], C = gatC[L], sh = gatSh[L];
        float* h   = buf2;
        float* out = buf1;
        gemm(gatIn, aw[L], h, N, K, F, nullptr, 0, wsW, wsA, stream);
        k_gat_logits<<<cdiv(N3, TPB), TPB, 0, stream>>>(h, as_[L], ad_[L], als, ald, N, C);
        k_gat_selfmax<<<cdiv(N3, TPB), TPB, 0, stream>>>(als, ald, mmax, (int)N3);
        k_gat_edgemax<<<cdiv((long long)E * 3, TPB), TPB, 0, stream>>>(src, dst, als, ald, mmax, E);
        k_gat_wsinit<<<cdiv(N3, TPB), TPB, 0, stream>>>(als, ald, mmax, denom, (int)N3);
        k_gat_outinit<<<cdiv((long long)N * F, TPB), TPB, 0, stream>>>(h, denom, out, N, F, sh);
        k_gat_edge<<<cdiv((long long)E * 32, TPB), TPB, 0, stream>>>(src, dst, als, ald, mmax, denom, h, out, E, F, sh);
        k_gat_final<<<cdiv((long long)N * F, TPB), TPB, 0, stream>>>(
            out, denom, ab_[L], bn[3 + L][0], bn[3 + L][1], bn[3 + L][2], bn[3 + L][3], N, F, sh);
        gatIn = out;
    }
    k_pool<<<cdiv((long long)N * 384, TPB), TPB, 0, stream>>>(buf1, batch, cat, N, 384, 512, 128);

    // =================== MLP head ===================
    gemm(cat, f1w, fc1, G, 512, 256, f1b, 1, wsW, wsA, stream);
    gemm(fc1, f2w, fc2, G, 256, 64, f2b, 1, wsW, wsA, stream);
    gemm(fc2, f3w, (float*)d_out, G, 64, 1, f3b, 1, wsW, wsA, stream);
}
